// StackedRNN_18184891531721
// MI455X (gfx1250) — compile-verified, matrix operand order checked by reading, not run on hardware
//
#include <hip/hip_runtime.h>
#include <hip/hip_bf16.h>
#include <math.h>

// Problem constants (match reference)
#define BB   4
#define SS   2048
#define HH   1024
#define LL   4
#define MM   (BB * SS)      // 8192 rows
#define CHUNKS 32
#define SCLEN  (SS / CHUNKS) // 64
#define BHCH   (BB * HH)     // 4096 channels

typedef __attribute__((ext_vector_type(16))) __bf16 v16bf;
typedef __attribute__((ext_vector_type(8)))  __bf16 v8bf;
typedef __attribute__((ext_vector_type(8)))  float  v8f;

// ---------------------------------------------------------------------------
// bf16 fragment loader for V_WMMA_F32_16X16X32_BF16 (ISA 7.12.2 layout).
// A (16x32, MxK): lane L<16 -> row M=L, K = {0..7} U {16..23};
//                 lane L>=16 -> row M=L-16, K = {8..15} U {24..31}.
// B (32x16, KxN) mirrors this with N per lane. Caller pre-offsets the row
// pointer by (hi*8) so both halves are contiguous 16-byte runs.
// ---------------------------------------------------------------------------
__device__ __forceinline__ v16bf load_frag_b16(const __bf16* p) {
  v8bf lo = *(const v8bf*)(p);
  v8bf hi = *(const v8bf*)(p + 16);
  v16bf f;
#pragma unroll
  for (int i = 0; i < 8; ++i) { f[i] = lo[i]; f[i + 8] = hi[i]; }
  return f;
}

// ---------------------------------------------------------------------------
// Gates GEMM: Z = X*Wz^T + bz ; Ht = X*Wh^T + bh  (shared A fragment)
// epilogue: a = clamp(1-sigmoid(Z)), bvec = sigmoid(Z)*tanh(Ht)
// X: (Mdim,K) bf16 row-major, Wz/Wh: (N,K) bf16 row-major.
// One wave -> a 16(M) x 32(N) strip of BOTH outputs: 4 accumulators.
// Software-pipelined: next K-step fragments are loaded before current WMMAs,
// so the pre-WMMA waits tolerate in-flight loads instead of loadcnt==0.
// ---------------------------------------------------------------------------
__global__ __launch_bounds__(256) void gates_gemm_wmma(
    const __bf16* __restrict__ X,
    const __bf16* __restrict__ Wz,
    const __bf16* __restrict__ Wh,
    const float* __restrict__ bz, const float* __restrict__ bh,
    float* __restrict__ avec, float* __restrict__ bvec,
    int Mdim, int N, int K)
{
  const int wave    = blockIdx.x * (blockDim.x >> 5) + (threadIdx.x >> 5);
  const int nStrips = N >> 5;                 // strips of 32 columns
  const int mTile   = wave / nStrips;
  const int strip   = wave % nStrips;
  if (mTile >= (Mdim >> 4)) return;

  const int lane = threadIdx.x & 31;
  const int lm   = lane & 15;
  const int hi8  = (lane >> 4) * 8;

  const __bf16* arow  = X  + (size_t)(mTile * 16 + lm) * K + hi8;
  const __bf16* zrow0 = Wz + (size_t)(strip * 32 + lm) * K + hi8;
  const __bf16* zrow1 = zrow0 + (size_t)16 * K;
  const __bf16* hrow0 = Wh + (size_t)(strip * 32 + lm) * K + hi8;
  const __bf16* hrow1 = hrow0 + (size_t)16 * K;

  v8f accZ0 = {}, accZ1 = {}, accH0 = {}, accH1 = {};

  // prologue: load K-step 0
  v16bf af = load_frag_b16(arow);
  v16bf z0 = load_frag_b16(zrow0);
  v16bf z1 = load_frag_b16(zrow1);
  v16bf h0 = load_frag_b16(hrow0);
  v16bf h1 = load_frag_b16(hrow1);

  for (int k = 0; k < K - 32; k += 32) {
    const int kn = k + 32;
    if (kn + 96 < K) {
      __builtin_prefetch(arow  + kn + 96, 0, 3);
      __builtin_prefetch(zrow0 + kn + 96, 0, 3);
      __builtin_prefetch(zrow1 + kn + 96, 0, 3);
      __builtin_prefetch(hrow0 + kn + 96, 0, 3);
      __builtin_prefetch(hrow1 + kn + 96, 0, 3);
    }
    // issue next-iteration loads first (double buffer)
    v16bf afn = load_frag_b16(arow  + kn);
    v16bf z0n = load_frag_b16(zrow0 + kn);
    v16bf z1n = load_frag_b16(zrow1 + kn);
    v16bf h0n = load_frag_b16(hrow0 + kn);
    v16bf h1n = load_frag_b16(hrow1 + kn);
    // compute on current fragments while next loads are in flight
    accZ0 = __builtin_amdgcn_wmma_f32_16x16x32_bf16(false, af, false, z0,
                                                    (short)0, accZ0, false, false);
    accZ1 = __builtin_amdgcn_wmma_f32_16x16x32_bf16(false, af, false, z1,
                                                    (short)0, accZ1, false, false);
    accH0 = __builtin_amdgcn_wmma_f32_16x16x32_bf16(false, af, false, h0,
                                                    (short)0, accH0, false, false);
    accH1 = __builtin_amdgcn_wmma_f32_16x16x32_bf16(false, af, false, h1,
                                                    (short)0, accH1, false, false);
    af = afn; z0 = z0n; z1 = z1n; h0 = h0n; h1 = h1n;
  }
  // epilogue K-step
  accZ0 = __builtin_amdgcn_wmma_f32_16x16x32_bf16(false, af, false, z0,
                                                  (short)0, accZ0, false, false);
  accZ1 = __builtin_amdgcn_wmma_f32_16x16x32_bf16(false, af, false, z1,
                                                  (short)0, accZ1, false, false);
  accH0 = __builtin_amdgcn_wmma_f32_16x16x32_bf16(false, af, false, h0,
                                                  (short)0, accH0, false, false);
  accH1 = __builtin_amdgcn_wmma_f32_16x16x32_bf16(false, af, false, h1,
                                                  (short)0, accH1, false, false);

#pragma unroll
  for (int sub = 0; sub < 2; ++sub) {
    const v8f* aZ = sub ? &accZ1 : &accZ0;
    const v8f* aH = sub ? &accH1 : &accH0;
    const int   col = strip * 32 + sub * 16 + lm;
    const float zb  = bz[col];
    const float hb  = bh[col];
#pragma unroll
    for (int r = 0; r < 8; ++r) {
      const int    row = mTile * 16 + r + hi8;  // C layout: lanes>=16 hold M=r+8
      const size_t off = (size_t)row * N + col;
      const float z = 1.0f / (1.0f + __expf(-((*aZ)[r] + zb)));
      const float t = tanhf((*aH)[r] + hb);
      float a = 1.0f - z;
      a = fminf(fmaxf(a, 1e-8f), 1.0f - 1e-8f);
      avec[off] = a;
      bvec[off] = z * t;
    }
  }
}

// ---------------------------------------------------------------------------
// Fusion GEMM: out = Comb * W^T + bias   (Comb: (Mdim,K) bf16, W: (N,K) bf16)
// One wave -> a 16(M) x 64(N) strip: 4 accumulators share one A fragment.
// Software-pipelined like the gates kernel.
// ---------------------------------------------------------------------------
__global__ __launch_bounds__(256) void fusion_gemm_wmma(
    const __bf16* __restrict__ X,
    const __bf16* __restrict__ W,
    const float* __restrict__ bias,
    float* __restrict__ out,
    int Mdim, int N, int K)
{
  const int wave    = blockIdx.x * (blockDim.x >> 5) + (threadIdx.x >> 5);
  const int nStrips = N >> 6;                 // strips of 64 columns
  const int mTile   = wave / nStrips;
  const int strip   = wave % nStrips;
  if (mTile >= (Mdim >> 4)) return;

  const int lane = threadIdx.x & 31;
  const int lm   = lane & 15;
  const int hi8  = (lane >> 4) * 8;

  const __bf16* arow = X + (size_t)(mTile * 16 + lm) * K + hi8;
  const __bf16* wrow[4];
#pragma unroll
  for (int u = 0; u < 4; ++u)
    wrow[u] = W + (size_t)(strip * 64 + u * 16 + lm) * K + hi8;

  v8f acc[4] = {{}, {}, {}, {}};

  v16bf af = load_frag_b16(arow);
  v16bf wf[4];
#pragma unroll
  for (int u = 0; u < 4; ++u) wf[u] = load_frag_b16(wrow[u]);

  for (int k = 0; k < K - 32; k += 32) {
    const int kn = k + 32;
    if (kn + 96 < K) {
      __builtin_prefetch(arow + kn + 96, 0, 3);
#pragma unroll
      for (int u = 0; u < 4; ++u) __builtin_prefetch(wrow[u] + kn + 96, 0, 3);
    }
    v16bf afn = load_frag_b16(arow + kn);
    v16bf wfn[4];
#pragma unroll
    for (int u = 0; u < 4; ++u) wfn[u] = load_frag_b16(wrow[u] + kn);
#pragma unroll
    for (int u = 0; u < 4; ++u)
      acc[u] = __builtin_amdgcn_wmma_f32_16x16x32_bf16(false, af, false, wf[u],
                                                       (short)0, acc[u], false, false);
    af = afn;
#pragma unroll
    for (int u = 0; u < 4; ++u) wf[u] = wfn[u];
  }
#pragma unroll
  for (int u = 0; u < 4; ++u)
    acc[u] = __builtin_amdgcn_wmma_f32_16x16x32_bf16(false, af, false, wf[u],
                                                     (short)0, acc[u], false, false);

#pragma unroll
  for (int u = 0; u < 4; ++u) {
    const int   col = strip * 64 + u * 16 + lm;
    const float bv  = bias[col];
#pragma unroll
    for (int r = 0; r < 8; ++r) {
      const int row = mTile * 16 + r + hi8;
      out[(size_t)row * N + col] = acc[u][r] + bv;
    }
  }
}

// ---------------------------------------------------------------------------
// f32 -> bf16 conversion (generic flat)
// ---------------------------------------------------------------------------
__global__ __launch_bounds__(256) void conv_f32_bf16(
    const float* __restrict__ src, __bf16* __restrict__ dst, size_t n)
{
  size_t i = (size_t)blockIdx.x * blockDim.x + threadIdx.x;
  if (i < n) dst[i] = (__bf16)src[i];
}

// f32 -> bf16 for (B,S,H) activations, optional time flip (bwd direction)
__global__ __launch_bounds__(256) void conv_act_bf16(
    const float* __restrict__ src, __bf16* __restrict__ dst, int flip)
{
  size_t i = (size_t)blockIdx.x * blockDim.x + threadIdx.x;
  if (i >= (size_t)MM * HH) return;
  if (!flip) { dst[i] = (__bf16)src[i]; return; }
  const int    c  = (int)(i % HH);
  const size_t bs = i / HH;
  const int    s  = (int)(bs % SS);
  const int    b  = (int)(bs / SS);
  dst[i] = (__bf16)src[((size_t)b * SS + (SS - 1 - s)) * HH + c];
}

// ---------------------------------------------------------------------------
// Chunked linear scan h_t = a_t*h_{t-1} + b_t over axis S (per (b,h) channel)
// Pass 1: local scan per chunk (h_local in place over bvec; record chunk
//         a-product and chunk-final h).
// Pass 2: sequential carry combine per channel over 32 chunks.
// Pass 3: fixup  h_t += prefixprod(a)*carry_in.
// ---------------------------------------------------------------------------
__global__ __launch_bounds__(256) void scan_local(
    const float* __restrict__ a, float* __restrict__ bv,
    float* __restrict__ aprod, float* __restrict__ hend)
{
  const int idx = blockIdx.x * blockDim.x + threadIdx.x;
  if (idx >= CHUNKS * BHCH) return;
  const int ch = idx % BHCH;
  const int j  = idx / BHCH;
  const int b  = ch / HH;
  const int c  = ch % HH;
  size_t o = ((size_t)b * SS + (size_t)j * SCLEN) * HH + c;
  float h = 0.0f, p = 1.0f;
#pragma unroll 4
  for (int t = 0; t < SCLEN; ++t, o += HH) {
    const float av = a[o];
    h = __builtin_fmaf(av, h, bv[o]);
    p *= av;
    bv[o] = h;
  }
  aprod[idx] = p;
  hend[idx]  = h;
}

__global__ __launch_bounds__(256) void scan_carry(
    float* __restrict__ aprod /* in: prod, out: carry_in */,
    const float* __restrict__ hend)
{
  const int ch = blockIdx.x * blockDim.x + threadIdx.x;
  if (ch >= BHCH) return;
  float carry = 0.0f;
#pragma unroll
  for (int j = 0; j < CHUNKS; ++j) {
    const int o  = j * BHCH + ch;
    const float ap = aprod[o];
    const float he = hend[o];
    aprod[o] = carry;                       // carry_in for chunk j
    carry = __builtin_fmaf(ap, carry, he);  // carry_out
  }
}

__global__ __launch_bounds__(256) void scan_apply(
    const float* __restrict__ a, float* __restrict__ bv,
    const float* __restrict__ carry_in)
{
  const int idx = blockIdx.x * blockDim.x + threadIdx.x;
  if (idx >= CHUNKS * BHCH) return;
  const float cin = carry_in[idx];
  if (cin == 0.0f) return;                  // chunk 0 and exact zeros: no-op
  const int ch = idx % BHCH;
  const int j  = idx / BHCH;
  const int b  = ch / HH;
  const int c  = ch % HH;
  size_t o = ((size_t)b * SS + (size_t)j * SCLEN) * HH + c;
  float p = 1.0f;
#pragma unroll 4
  for (int t = 0; t < SCLEN; ++t, o += HH) {
    p *= a[o];
    bv[o] = __builtin_fmaf(p, cin, bv[o]);
  }
}

// ---------------------------------------------------------------------------
// residual add + LayerNorm: one 256-thread block per (b,s) row, H=1024.
// ---------------------------------------------------------------------------
__global__ __launch_bounds__(256) void resid_layernorm(
    const float* __restrict__ hnew, const float* __restrict__ hprev,
    float* __restrict__ outp,
    const float* __restrict__ gamma, const float* __restrict__ beta,
    int use_resid)
{
  const size_t base = (size_t)blockIdx.x * HH;
  float v[4];
  float s = 0.0f;
#pragma unroll
  for (int j = 0; j < 4; ++j) {
    const int c = threadIdx.x + j * 256;
    float xv = hnew[base + c];
    if (use_resid) xv += hprev[base + c];
    v[j] = xv;
    s += xv;
  }
  __shared__ float red[256];
  red[threadIdx.x] = s;
  __syncthreads();
  for (int st = 128; st > 0; st >>= 1) {
    if (threadIdx.x < st) red[threadIdx.x] += red[threadIdx.x + st];
    __syncthreads();
  }
  const float mu = red[0] * (1.0f / HH);
  __syncthreads();
  float s2 = 0.0f;
#pragma unroll
  for (int j = 0; j < 4; ++j) { const float d = v[j] - mu; s2 += d * d; }
  red[threadIdx.x] = s2;
  __syncthreads();
  for (int st = 128; st > 0; st >>= 1) {
    if (threadIdx.x < st) red[threadIdx.x] += red[threadIdx.x + st];
    __syncthreads();
  }
  const float inv = rsqrtf(red[0] * (1.0f / HH) + 1e-5f);
#pragma unroll
  for (int j = 0; j < 4; ++j) {
    const int c = threadIdx.x + j * 256;
    outp[base + c] = (v[j] - mu) * inv * gamma[c] + beta[c];
  }
}

// ---------------------------------------------------------------------------
// concat [h_fwd, flip(h_bwd)] -> bf16 (B,S,2H)
// ---------------------------------------------------------------------------
__global__ __launch_bounds__(256) void concat_bf16(
    const float* __restrict__ hf, const float* __restrict__ hb,
    __bf16* __restrict__ comb)
{
  size_t i = (size_t)blockIdx.x * blockDim.x + threadIdx.x;
  if (i >= (size_t)MM * 2 * HH) return;
  const int    c  = (int)(i % (2 * HH));
  const size_t bs = i / (2 * HH);
  const int    s  = (int)(bs % SS);
  const int    b  = (int)(bs / SS);
  float v;
  if (c < HH) v = hf[((size_t)b * SS + s) * HH + c];
  else        v = hb[((size_t)b * SS + (SS - 1 - s)) * HH + (c - HH)];
  comb[i] = (__bf16)v;
}

// ---------------------------------------------------------------------------
extern "C" void kernel_launch(void* const* d_in, const int* in_sizes, int n_in,
                              void* d_out, int out_size, void* d_ws, size_t ws_size,
                              hipStream_t stream) {
  const float* x        = (const float*)d_in[0];
  const float* fwd_Wz   = (const float*)d_in[1];
  const float* fwd_bz   = (const float*)d_in[2];
  const float* fwd_Wh   = (const float*)d_in[3];
  const float* fwd_bh   = (const float*)d_in[4];
  const float* bwd_Wz   = (const float*)d_in[5];
  const float* bwd_bz   = (const float*)d_in[6];
  const float* bwd_Wh   = (const float*)d_in[7];
  const float* bwd_bh   = (const float*)d_in[8];
  const float* fusion_W = (const float*)d_in[9];
  const float* fusion_b = (const float*)d_in[10];
  const float* gamma    = (const float*)d_in[11];
  const float* beta     = (const float*)d_in[12];
  float* out = (float*)d_out;

  // Workspace carve-out (~197 MB)
  uint8_t* p = (uint8_t*)d_ws;
  auto take = [&](size_t bytes) -> void* {
    void* r = (void*)p;
    p += (bytes + 255) & ~(size_t)255;
    return r;
  };
  __bf16* wbf   = (__bf16*)take((size_t)2 * HH * HH * sizeof(__bf16)); //  4 MB
  __bf16* xbf   = (__bf16*)take((size_t)MM * 2 * HH * sizeof(__bf16)); // 32 MB
  float*  avec  = (float*) take((size_t)MM * HH * sizeof(float));      // 32 MB
  float*  bvec  = (float*) take((size_t)MM * HH * sizeof(float));      // 32 MB
  float*  hbuf  = (float*) take((size_t)MM * HH * sizeof(float));      // 32 MB
  float*  hfwd  = (float*) take((size_t)MM * HH * sizeof(float));      // 32 MB
  float*  hbwd  = (float*) take((size_t)MM * HH * sizeof(float));      // 32 MB
  float*  aprod = (float*) take((size_t)CHUNKS * BHCH * sizeof(float));
  float*  hend  = (float*) take((size_t)CHUNKS * BHCH * sizeof(float));

  const int actBlocks   = (int)(((size_t)MM * HH) / 256);        // 32768
  const int wBlocks     = (HH * HH) / 256;                       //  4096
  const int gatesBlocks = ((MM / 16) * (HH / 32)) / 8;           //  2048 (8 waves/blk)
  const int fuseBlocks  = ((MM / 16) * (HH / 64)) / 8;           //  1024
  const int scanBlocks  = (CHUNKS * BHCH) / 256;                 //   512
  const int carryBlocks = BHCH / 256;                            //    16
  const int catBlocks   = (int)(((size_t)MM * 2 * HH) / 256);    // 65536

  for (int dir = 0; dir < 2; ++dir) {
    const float* Wz = dir ? bwd_Wz : fwd_Wz;
    const float* Wh = dir ? bwd_Wh : fwd_Wh;
    const float* bz = dir ? bwd_bz : fwd_bz;
    const float* bh = dir ? bwd_bh : fwd_bh;
    float* hout = dir ? hbwd : hfwd;

    // layer-0 input (bwd: time-flipped x)
    conv_act_bf16<<<actBlocks, 256, 0, stream>>>(x, xbf, dir);

    for (int i = 0; i < LL; ++i) {
      const size_t woff = (size_t)i * HH * HH;
      conv_f32_bf16<<<wBlocks, 256, 0, stream>>>(Wz + woff, wbf, (size_t)HH * HH);
      conv_f32_bf16<<<wBlocks, 256, 0, stream>>>(Wh + woff, wbf + (size_t)HH * HH,
                                                 (size_t)HH * HH);

      gates_gemm_wmma<<<gatesBlocks, 256, 0, stream>>>(
          xbf, wbf, wbf + (size_t)HH * HH, bz + i * HH, bh + i * HH,
          avec, bvec, MM, HH, HH);

      scan_local<<<scanBlocks, 256, 0, stream>>>(avec, bvec, aprod, hend);
      scan_carry<<<carryBlocks, 256, 0, stream>>>(aprod, hend);
      scan_apply<<<scanBlocks, 256, 0, stream>>>(avec, bvec, aprod);

      float* lnout = (i == LL - 1) ? hout : hbuf;
      resid_layernorm<<<MM, 256, 0, stream>>>(bvec, hbuf, lnout, gamma, beta,
                                              (i > 0) ? 1 : 0);
      if (i < LL - 1)
        conv_act_bf16<<<actBlocks, 256, 0, stream>>>(hbuf, xbf, 0);
    }
  }

  // fusion: concat -> bf16 GEMM (K = 2H) with bias
  concat_bf16<<<catBlocks, 256, 0, stream>>>(hfwd, hbwd, xbf);
  conv_f32_bf16<<<(HH * 2 * HH) / 256, 256, 0, stream>>>(fusion_W, wbf,
                                                         (size_t)HH * 2 * HH);
  fusion_gemm_wmma<<<fuseBlocks, 256, 0, stream>>>(xbf, wbf, fusion_b, out,
                                                   MM, HH, 2 * HH);
}
